// ARDBayesianLinear_22986664968527
// MI455X (gfx1250) — matrix operand order repaired and checked by reading, not compiled, not run on hardware
//
#include <hip/hip_runtime.h>
#include <hip/hip_bf16.h>

// ---------------- problem constants ----------------
#define IN_F   4096
#define OUT_F  4096
#define BATCH  8192
#define KEEP   0.9f

typedef __attribute__((ext_vector_type(16))) __bf16 v16bf;
typedef __attribute__((ext_vector_type(8)))  float  v8f;

__device__ __forceinline__ float sp(float v) {
    // numerically stable softplus
    return fmaxf(v, 0.0f) + log1pf(expf(-fabsf(v)));
}

__device__ __forceinline__ unsigned short f2bf(float f) {
    unsigned int u = __float_as_uint(f);
    u += 0x7FFFu + ((u >> 16) & 1u);      // round-to-nearest-even
    return (unsigned short)(u >> 16);
}

// ---------------- kernel 1: scales, bias sample, ard+bias KL ----------------
__global__ __launch_bounds__(256)
void prep_small_kernel(const float* __restrict__ bias_mu,
                       const float* __restrict__ bias_rho,
                       const float* __restrict__ ard_alpha,
                       const float* __restrict__ ard_beta,
                       const float* __restrict__ bias_noise,
                       float* __restrict__ ard_scale,
                       float* __restrict__ bias_out,
                       float* __restrict__ kl_out) {
    int i = blockIdx.x * 256 + threadIdx.x;   // 4096 total
    float kacc = 0.0f;
    if (i < IN_F) {
        float sa = sp(ard_alpha[i]);
        float sb = sp(ard_beta[i]);
        ard_scale[i] = sa * sb;
        float bs = sp(bias_rho[i]);
        float bm = bias_mu[i];
        bias_out[i] = bm + bs * bias_noise[i];
        kacc  = sa + sb - logf(sa) - logf(sb);                                // ard KL
        kacc += 0.5f * (2.0f * logf(bs) + 1.0f / (bs * bs) + bm * bm - 1.0f); // bias KL
    }
    __shared__ float red[256];
    red[threadIdx.x] = kacc;
    __syncthreads();
    for (int s = 128; s > 0; s >>= 1) {
        if (threadIdx.x < s) red[threadIdx.x] += red[threadIdx.x + s];
        __syncthreads();
    }
    if (threadIdx.x == 0) atomicAdd(kl_out, red[0]);
}

// ---------------- kernel 2: sampled weight -> bf16, weight KL ----------------
__global__ __launch_bounds__(256)
void prep_w_kernel(const float4* __restrict__ mu4,
                   const float4* __restrict__ rho4,
                   const float4* __restrict__ noise4,
                   unsigned short* __restrict__ wbf,
                   float* __restrict__ kl_out) {
    int idx = blockIdx.x * 256 + threadIdx.x;   // OUT_F*IN_F/4 = 4194304
    float4 m = mu4[idx];
    float4 r = rho4[idx];
    float4 n = noise4[idx];
    float kacc = 0.0f;
    ushort4 o;
    {
        float s = sp(r.x); o.x = f2bf(m.x + s * n.x);
        kacc += 0.5f * (2.0f * logf(s) + 1.0f / (s * s) + m.x * m.x - 1.0f);
    }
    {
        float s = sp(r.y); o.y = f2bf(m.y + s * n.y);
        kacc += 0.5f * (2.0f * logf(s) + 1.0f / (s * s) + m.y * m.y - 1.0f);
    }
    {
        float s = sp(r.z); o.z = f2bf(m.z + s * n.z);
        kacc += 0.5f * (2.0f * logf(s) + 1.0f / (s * s) + m.z * m.z - 1.0f);
    }
    {
        float s = sp(r.w); o.w = f2bf(m.w + s * n.w);
        kacc += 0.5f * (2.0f * logf(s) + 1.0f / (s * s) + m.w * m.w - 1.0f);
    }
    *(ushort4*)(wbf + (size_t)idx * 4) = o;

    __shared__ float red[256];
    red[threadIdx.x] = kacc;
    __syncthreads();
    for (int s = 128; s > 0; s >>= 1) {
        if (threadIdx.x < s) red[threadIdx.x] += red[threadIdx.x + s];
        __syncthreads();
    }
    if (threadIdx.x == 0) atomicAdd(kl_out, red[0]);
}

// ---------------- kernel 3: x * dropout * ard_scale -> bf16 ----------------
__global__ __launch_bounds__(256)
void prep_x_kernel(const float4* __restrict__ x4,
                   const float4* __restrict__ u4,
                   const float4* __restrict__ scale4,
                   unsigned short* __restrict__ xbf) {
    int idx = blockIdx.x * 256 + threadIdx.x;   // BATCH*IN_F/4 = 8388608
    float4 x = x4[idx];
    float4 u = u4[idx];
    float4 s = scale4[idx & (IN_F / 4 - 1)];
    const float inv = 1.0f / KEEP;
    ushort4 o;
    o.x = f2bf(x.x * ((u.x < KEEP) ? inv : 0.0f) * s.x);
    o.y = f2bf(x.y * ((u.y < KEEP) ? inv : 0.0f) * s.y);
    o.z = f2bf(x.z * ((u.z < KEEP) ? inv : 0.0f) * s.z);
    o.w = f2bf(x.w * ((u.w < KEEP) ? inv : 0.0f) * s.w);
    *(ushort4*)(xbf + (size_t)idx * 4) = o;
}

// ---------------- kernel 4: bf16 WMMA GEMM  C = Xd * W^T + bias ----------------
// Tile: 128(M) x 128(N) x 32(K). 256 threads = 8 waves in a 2(M) x 4(N) grid,
// each wave computes 64x32 via 4x2 accumulators of v_wmma_f32_16x16x32_bf16.
// Tiles staged with GLOBAL_LOAD_ASYNC_TO_LDS_B128 (ASYNCcnt), two-stage
// software pipeline with compile-time buffer indices (no phi/mov churn).
#define TM 128
#define TN 128
#define TK 32
#define LDSK 40   // padded bf16 row stride: 80B, multiple of 16B, conflict-free frags

#define HAS_ASYNC_LDS __has_builtin(__builtin_amdgcn_global_load_async_to_lds_b128)

#if HAS_ASYNC_LDS
// builtin signature (probe-confirmed): (v4i AS1*, v4i AS3*, imm int, imm int)
typedef int v4i_vs __attribute__((vector_size(16)));
#define AS1Q(p) ((__attribute__((address_space(1))) v4i_vs*)(p))
#define AS3Q(p) ((__attribute__((address_space(3))) v4i_vs*)(p))
__device__ __forceinline__ void wait_async0() {
#if __has_builtin(__builtin_amdgcn_s_wait_asynccnt)
    __builtin_amdgcn_s_wait_asynccnt(0);
#else
    asm volatile("s_wait_asynccnt 0x0" ::: "memory");
#endif
}
#endif

union FragU { uint4 q[2]; v16bf v; };

__global__ __launch_bounds__(256)
void gemm_bf16_kernel(const unsigned short* __restrict__ A,   // [BATCH, IN_F] bf16
                      const unsigned short* __restrict__ B,   // [OUT_F, IN_F] bf16
                      const float* __restrict__ bias,         // [OUT_F]
                      float* __restrict__ C) {                // [BATCH, OUT_F]
    __shared__ unsigned short lA[2][TM * LDSK];
    __shared__ unsigned short lB[2][TN * LDSK];

    const int tid  = threadIdx.x;
    const int lane = tid & 31;
    const int wave = tid >> 5;
    const int wm   = wave >> 2;      // 0..1
    const int wn   = wave & 3;       // 0..3
    const int lr   = lane & 15;
    const int hi   = lane >> 4;      // 0 or 1

    const int m_block = blockIdx.y * TM;
    const int n_block = blockIdx.x * TN;

    // tile staging: thread -> (row, chunk); chunk = 8 bf16 = 16B; conflict-free
    const int srow  = tid & 127;
    const int c0    = tid >> 7;      // 0 or 1
    const int c1    = c0 + 2;

    const unsigned short* aptr = A + (size_t)(m_block + srow) * IN_F;
    const unsigned short* bptr = B + (size_t)(n_block + srow) * IN_F;
    unsigned short* lArow = &lA[0][srow * LDSK];
    unsigned short* lBrow = &lB[0][srow * LDSK];

#if HAS_ASYNC_LDS
#define ISSUE_TILE(k0, buf)                                                                 \
    do {                                                                                    \
        __builtin_amdgcn_global_load_async_to_lds_b128(                                     \
            AS1Q(aptr + (k0) + c0 * 8), AS3Q(lArow + (buf) * (TM * LDSK) + c0 * 8), 0, 0);  \
        __builtin_amdgcn_global_load_async_to_lds_b128(                                     \
            AS1Q(aptr + (k0) + c1 * 8), AS3Q(lArow + (buf) * (TM * LDSK) + c1 * 8), 0, 0);  \
        __builtin_amdgcn_global_load_async_to_lds_b128(                                     \
            AS1Q(bptr + (k0) + c0 * 8), AS3Q(lBrow + (buf) * (TN * LDSK) + c0 * 8), 0, 0);  \
        __builtin_amdgcn_global_load_async_to_lds_b128(                                     \
            AS1Q(bptr + (k0) + c1 * 8), AS3Q(lBrow + (buf) * (TN * LDSK) + c1 * 8), 0, 0);  \
    } while (0)
#define COMMIT_TILE(buf)     do { } while (0)
#define WAIT_TILE()          do { wait_async0(); } while (0)
#else
    // fallback: named scalar staging registers (no arrays/lambdas -> no spills)
    uint4 ra0, ra1, rb0, rb1;
#define ISSUE_TILE(k0, buf)                              \
    do {                                                 \
        ra0 = *(const uint4*)(aptr + (k0) + c0 * 8);     \
        ra1 = *(const uint4*)(aptr + (k0) + c1 * 8);     \
        rb0 = *(const uint4*)(bptr + (k0) + c0 * 8);     \
        rb1 = *(const uint4*)(bptr + (k0) + c1 * 8);     \
    } while (0)
#define COMMIT_TILE(buf)                                           \
    do {                                                           \
        *(uint4*)(lArow + (buf) * (TM * LDSK) + c0 * 8) = ra0;     \
        *(uint4*)(lArow + (buf) * (TM * LDSK) + c1 * 8) = ra1;     \
        *(uint4*)(lBrow + (buf) * (TN * LDSK) + c0 * 8) = rb0;     \
        *(uint4*)(lBrow + (buf) * (TN * LDSK) + c1 * 8) = rb1;     \
    } while (0)
#define WAIT_TILE()          do { } while (0)
#endif

    // ISA A layout: lanes<16 hold K{0..7,16..23}, lanes>=16 K{8..15,24..31}
#define LOAD_A(fr, buf, ai)                                                        \
    do {                                                                           \
        const unsigned short* p = &lA[buf][(wm * 64 + (ai) * 16 + lr) * LDSK];     \
        fr.q[0] = *(const uint4*)(p + hi * 8);                                     \
        fr.q[1] = *(const uint4*)(p + 16 + hi * 8);                                \
    } while (0)
    // ISA B layout: lanes<16 hold K{0..7,16..23}, lanes>=16 K{8..15,24..31}
#define LOAD_B(fr, buf, bi)                                                        \
    do {                                                                           \
        const unsigned short* p = &lB[buf][(wn * 32 + (bi) * 16 + lr) * LDSK];     \
        fr.q[0] = *(const uint4*)(p + hi * 8);                                     \
        fr.q[1] = *(const uint4*)(p + 16 + hi * 8);                                \
    } while (0)
#define WMMA(c, a, b)                                                              \
    c = __builtin_amdgcn_wmma_f32_16x16x32_bf16(false, (a).v, false, (b).v,        \
                                                (short)0, c, false, false)
#define COMPUTE(buf)                                                               \
    do {                                                                           \
        FragU fa0, fa1, fa2, fa3, fb0, fb1;                                        \
        LOAD_A(fa0, buf, 0); LOAD_A(fa1, buf, 1);                                  \
        LOAD_A(fa2, buf, 2); LOAD_A(fa3, buf, 3);                                  \
        LOAD_B(fb0, buf, 0); LOAD_B(fb1, buf, 1);                                  \
        WMMA(c00, fa0, fb0); WMMA(c01, fa0, fb1);                                  \
        WMMA(c10, fa1, fb0); WMMA(c11, fa1, fb1);                                  \
        WMMA(c20, fa2, fb0); WMMA(c21, fa2, fb1);                                  \
        WMMA(c30, fa3, fb0); WMMA(c31, fa3, fb1);                                  \
    } while (0)

    v8f c00 = {}, c01 = {}, c10 = {}, c11 = {};
    v8f c20 = {}, c21 = {}, c30 = {}, c31 = {};

    ISSUE_TILE(0, 0);
    COMMIT_TILE(0);
    WAIT_TILE();
    __syncthreads();

    const int NK = IN_F / TK;  // 128, even
    for (int kt = 0; kt < NK; kt += 2) {
        // stage 0: compute buf0, fetch kt+1 into buf1
        ISSUE_TILE((kt + 1) * TK, 1);
        COMPUTE(0);
        COMMIT_TILE(1);
        WAIT_TILE();
        __syncthreads();
        // stage 1: compute buf1, fetch kt+2 into buf0
        if (kt + 2 < NK) ISSUE_TILE((kt + 2) * TK, 0);
        COMPUTE(1);
        if (kt + 2 < NK) COMMIT_TILE(0);
        WAIT_TILE();
        __syncthreads();
    }

    // epilogue: C/D layout -> VGPR r holds M = r + (lane<16 ? 0 : 8), N = lane%16
#define STORE_ACC(ai, bi, c)                                                   \
    do {                                                                       \
        const int rbase = m_block + wm * 64 + (ai) * 16 + hi * 8;              \
        const int col   = n_block + wn * 32 + (bi) * 16 + lr;                  \
        const float bv  = bias[col];                                           \
        float* outp = C + (size_t)rbase * OUT_F + col;                         \
        _Pragma("unroll")                                                      \
        for (int r = 0; r < 8; ++r)                                            \
            outp[(size_t)r * OUT_F] = c[r] + bv;                               \
    } while (0)

    STORE_ACC(0, 0, c00); STORE_ACC(0, 1, c01);
    STORE_ACC(1, 0, c10); STORE_ACC(1, 1, c11);
    STORE_ACC(2, 0, c20); STORE_ACC(2, 1, c21);
    STORE_ACC(3, 0, c30); STORE_ACC(3, 1, c31);
}

// ---------------- host-side launch ----------------
extern "C" void kernel_launch(void* const* d_in, const int* in_sizes, int n_in,
                              void* d_out, int out_size, void* d_ws, size_t ws_size,
                              hipStream_t stream) {
    const float* x            = (const float*)d_in[0];
    const float* weight_mu    = (const float*)d_in[1];
    const float* weight_rho   = (const float*)d_in[2];
    const float* bias_mu      = (const float*)d_in[3];
    const float* bias_rho     = (const float*)d_in[4];
    const float* ard_alpha    = (const float*)d_in[5];
    const float* ard_beta     = (const float*)d_in[6];
    const float* weight_noise = (const float*)d_in[7];
    const float* bias_noise   = (const float*)d_in[8];
    const float* dropout_u    = (const float*)d_in[9];

    float* out = (float*)d_out;
    float* kl_slot = out + (out_size - 1);   // last element = KL scalar

    // workspace layout
    char* ws = (char*)d_ws;
    unsigned short* wbf  = (unsigned short*)ws;                                   // 32 MB
    unsigned short* xbf  = (unsigned short*)(ws + (size_t)OUT_F * IN_F * 2);      // 64 MB
    float* ard_scale     = (float*)(ws + (size_t)OUT_F * IN_F * 2
                                       + (size_t)BATCH * IN_F * 2);
    float* bias_s        = ard_scale + IN_F;

    // zero KL accumulator (re-zeroed every call -> deterministic under graph replay)
    (void)hipMemsetAsync(kl_slot, 0, sizeof(float), stream);

    // 1) scales + bias sample + ard/bias KL
    prep_small_kernel<<<IN_F / 256, 256, 0, stream>>>(
        bias_mu, bias_rho, ard_alpha, ard_beta, bias_noise,
        ard_scale, bias_s, kl_slot);

    // 2) sampled weight -> bf16 + weight KL
    prep_w_kernel<<<(OUT_F * IN_F / 4) / 256, 256, 0, stream>>>(
        (const float4*)weight_mu, (const float4*)weight_rho,
        (const float4*)weight_noise, wbf, kl_slot);

    // 3) dropout * ard * x -> bf16
    prep_x_kernel<<<(BATCH * IN_F / 4) / 256, 256, 0, stream>>>(
        (const float4*)x, (const float4*)dropout_u,
        (const float4*)ard_scale, xbf);

    // 4) WMMA GEMM + bias
    dim3 grid(OUT_F / TN, BATCH / TM);   // (32, 64)
    gemm_bf16_kernel<<<grid, 256, 0, stream>>>(xbf, wbf, bias_s, out);
}
